// RNN_63282048139410
// MI455X (gfx1250) — matrix operand order repaired and checked by reading, not compiled, not run on hardware
//
#include <hip/hip_runtime.h>
#include <hip/hip_bf16.h>
#include <math.h>

// ---------------------------------------------------------------------------
// MI455X (gfx1250) RNN forward: bf16 WMMA + async global->LDS staged W_rec
// ---------------------------------------------------------------------------

typedef __attribute__((ext_vector_type(8)))  float   v8f;
typedef __attribute__((ext_vector_type(8)))  __bf16  v8bf;
typedef __attribute__((ext_vector_type(16))) __bf16  v16bf;

// GCC-style int vectors matching the async-copy builtin prototypes.
typedef int gv4i __attribute__((vector_size(4 * sizeof(int))));
typedef int gv2i __attribute__((vector_size(2 * sizeof(int))));

#define RULE_START 85
#define N_RULE     20
#define N_IN       (RULE_START + N_RULE)   // 105
#define K_IN       128                     // padded K for input GEMM
#define N_OUT      33
#define N_OUT_PAD  48
#define N_RNN      1024
#define T_STEPS    256
#define BATCH      256

#define AS1 __attribute__((address_space(1)))
#define AS3 __attribute__((address_space(3)))

#if __has_builtin(__builtin_amdgcn_global_load_async_to_lds_b128) && \
    __has_builtin(__builtin_amdgcn_global_load_async_to_lds_b64)
#define ASYNC_COPY_OK 1
#else
#define ASYNC_COPY_OK 0
#endif

static __device__ __forceinline__ v16bf cat8(v8bf lo, v8bf hi) {
  return __builtin_shufflevector(lo, hi, 0,1,2,3,4,5,6,7,8,9,10,11,12,13,14,15);
}

static __device__ __forceinline__ v8f wmma_bf16(v16bf a, v16bf b, v8f c) {
  // (neg_a, A, neg_b, B, c_mod, C, reuse_a, reuse_b)
  return __builtin_amdgcn_wmma_f32_16x16x32_bf16(false, a, false, b,
                                                 (short)0, c, false, false);
}

static __device__ __forceinline__ float softplusf(float x) {
  // numerically stable: max(x,0) + log1p(exp(-|x|))
  return fmaxf(x, 0.0f) + log1pf(__expf(-fabsf(x)));
}

// --- async global->LDS copy helpers (fallback: sync register staging) ------

static __device__ __forceinline__ void copy_g2l_b128(void* ldsp, const void* gp) {
#if ASYNC_COPY_OK
  __builtin_amdgcn_global_load_async_to_lds_b128((AS1 gv4i*)gp, (AS3 gv4i*)ldsp, 0, 0);
#else
  *(v8bf*)ldsp = *(const v8bf*)gp;
#endif
}

static __device__ __forceinline__ void copy_g2l_b64(void* ldsp, const void* gp) {
#if ASYNC_COPY_OK
  __builtin_amdgcn_global_load_async_to_lds_b64((AS1 gv2i*)gp, (AS3 gv2i*)ldsp, 0, 0);
#else
  *(unsigned long long*)ldsp = *(const unsigned long long*)gp;
#endif
}

static __device__ __forceinline__ void wait_async3() {
#if __has_builtin(__builtin_amdgcn_s_wait_asynccnt)
  __builtin_amdgcn_s_wait_asynccnt(3);
#else
  asm volatile("s_wait_asynccnt 0x3" ::: "memory");
#endif
}

static __device__ __forceinline__ void wait_async0() {
#if __has_builtin(__builtin_amdgcn_s_wait_asynccnt)
  __builtin_amdgcn_s_wait_asynccnt(0);
#else
  asm volatile("s_wait_asynccnt 0x0" ::: "memory");
#endif
}

// ------------------------------ prep kernels -------------------------------

__global__ void RNN_prep_x(const float* __restrict__ x, __bf16* __restrict__ xbf) {
  size_t idx = (size_t)blockIdx.x * blockDim.x + threadIdx.x;   // < T*B*K_IN
  int    k   = (int)(idx & (K_IN - 1));
  size_t row = idx >> 7;
  float v = (k < N_IN) ? x[row * N_IN + k] : 0.0f;
  xbf[idx] = (__bf16)v;
}

__global__ void RNN_prep_wrec(const float* __restrict__ w, __bf16* __restrict__ wbf) {
  size_t idx = (size_t)blockIdx.x * blockDim.x + threadIdx.x;   // < 1024*1024
  wbf[idx] = (__bf16)w[idx];
}

__global__ void RNN_prep_win(const float* __restrict__ w_sens,
                             const float* __restrict__ w_rule,
                             __bf16* __restrict__ win) {
  size_t idx = (size_t)blockIdx.x * blockDim.x + threadIdx.x;   // < 1024*K_IN
  int n = (int)(idx >> 7);
  int k = (int)(idx & (K_IN - 1));
  float v = 0.0f;
  if (k < RULE_START)      v = w_sens[(size_t)n * RULE_START + k];
  else if (k < N_IN)       v = w_rule[(size_t)n * N_RULE + (k - RULE_START)];
  win[idx] = (__bf16)v;
}

__global__ void RNN_prep_wout(const float* __restrict__ w, __bf16* __restrict__ wbf) {
  size_t idx = (size_t)blockIdx.x * blockDim.x + threadIdx.x;   // < 48*1024
  int n = (int)(idx >> 10);
  int k = (int)(idx & (N_RNN - 1));
  float v = (n < N_OUT) ? w[(size_t)n * N_RNN + k] : 0.0f;
  wbf[idx] = (__bf16)v;
}

__global__ void RNN_prep_misc(const float* __restrict__ b_sens,
                              const float* __restrict__ b_rec,
                              float* __restrict__ bias,
                              float* __restrict__ hf32,
                              __bf16* __restrict__ hbf0) {
  size_t idx = (size_t)blockIdx.x * blockDim.x + threadIdx.x;   // < 256*1024
  hf32[idx] = 0.0f;
  hbf0[idx] = (__bf16)0.0f;
  if (idx < N_RNN) bias[idx] = b_sens[idx] + b_rec[idx];
}

// ------------------------------- step kernel -------------------------------
// Grid: (N_RNN/128, BATCH/32) x 256 threads (8 waves).
// Wave owns a 16(M) x 32(N) region = two 16x16 accumulator tiles.
// Phase 2 stages W_rec (128x32) and h (32x32) tiles in LDS, double-buffered,
// via async global->LDS copies tracked with ASYNCcnt.

#define LDS_B_BYTES (128 * 64)   // 128 n-rows x 32 bf16 (64B row stride) = 8192
#define LDS_A_BYTES (32 * 64)    // 32 m-rows  x 32 bf16 (64B row stride) = 2048
#define NCHUNK      (N_RNN / 32) // 32 K-chunks

__global__ __launch_bounds__(256, 1)
void RNN_step_kernel(const __bf16* __restrict__ xbf,    // (T*B, 128)
                     const __bf16* __restrict__ wrec,   // (1024, 1024)
                     const __bf16* __restrict__ win,    // (1024, 128)
                     const float*  __restrict__ bias,   // (1024)  b_sens+b_rec
                     const float*  __restrict__ noise,  // (T, B, 1024)
                     float*        __restrict__ hf32,   // (256, 1024)
                     const __bf16* __restrict__ hprev,  // (256, 1024)
                     __bf16*       __restrict__ hcur,   // (256, 1024)
                     int t)
{
  __shared__ __attribute__((aligned(32)))
      unsigned char lds[2][LDS_B_BYTES + LDS_A_BYTES];   // 2 x 10240 B

  const int tid  = threadIdx.x;
  const int lane = tid & 31;
  const int wave = tid >> 5;
  const int l15  = lane & 15;
  const int ksel = lane >> 4;                 // 0 or 1: which K half this lane holds
  const int mt   = wave >> 2;                 // m-tile within block (0..1)
  const int ng   = wave & 3;                  // 32-wide n-group within block (0..3)
  const int bm0  = blockIdx.y * 32;
  const int bn0  = blockIdx.x * 128;
  const int base_m = bm0 + mt * 16;
  const int base_n = bn0 + ng * 32;
  const int mrow   = base_m + l15;

  v8f acc0 = {};
  v8f acc1 = {};

  // ---- Phase 1: x[t] @ W_in^T  (K = 128, W_in is 256KB, L2-hot) ----
  {
    const __bf16* arow  = xbf + ((size_t)t * BATCH + mrow) * K_IN;
    const __bf16* brow0 = win + (size_t)(base_n + l15) * K_IN;
    const __bf16* brow1 = win + (size_t)(base_n + 16 + l15) * K_IN;
#pragma unroll
    for (int kk = 0; kk < K_IN; kk += 32) {
      v8bf alo = *(const v8bf*)(arow + kk + ksel * 16);
      v8bf ahi = *(const v8bf*)(arow + kk + ksel * 16 + 8);
      v16bf a  = cat8(alo, ahi);
      v16bf b0 = *(const v16bf*)(brow0 + kk + ksel * 16);
      v16bf b1 = *(const v16bf*)(brow1 + kk + ksel * 16);
      acc0 = wmma_bf16(a, b0, acc0);
      acc1 = wmma_bf16(a, b1, acc1);
    }
  }

  // ---- Phase 2: h_{t-1} @ W_rec^T  (K = 1024), LDS double-buffered ----
  {
    // Per chunk each thread issues exactly 3 async copies (2xB128 + 1xB64),
    // so every wave has 3 ASYNCcnt entries per chunk.
    const int sB0 = tid;            // B slots: 512 x 16B; row = s>>2, seg = s&3
    const int sB1 = tid + 256;
    const int rA  = tid >> 3;       // A: 256 x 8B; row = tid>>3, seg = tid&7
    const int gA  = tid & 7;

    auto issue_chunk = [&](int kk, int buf) {
      unsigned char* LB = &lds[buf][0];
      unsigned char* LA = &lds[buf][LDS_B_BYTES];
      copy_g2l_b128(LB + (sB0 >> 2) * 64 + (sB0 & 3) * 16,
                    wrec + (size_t)(bn0 + (sB0 >> 2)) * N_RNN + kk + (sB0 & 3) * 8);
      copy_g2l_b128(LB + (sB1 >> 2) * 64 + (sB1 & 3) * 16,
                    wrec + (size_t)(bn0 + (sB1 >> 2)) * N_RNN + kk + (sB1 & 3) * 8);
      copy_g2l_b64(LA + rA * 64 + gA * 8,
                   hprev + (size_t)(bm0 + rA) * N_RNN + kk + gA * 4);
    };

    issue_chunk(0, 0);
    for (int c = 0; c < NCHUNK; ++c) {
      if (c + 1 < NCHUNK) {
        issue_chunk((c + 1) * 32, (c + 1) & 1);
        wait_async3();               // chunk c's 3 copies are complete
      } else {
        wait_async0();
      }
      __syncthreads();               // LDS writes visible to all waves

      const unsigned char* LB = &lds[c & 1][0];
      const unsigned char* LA = &lds[c & 1][LDS_B_BYTES];
      v8bf alo = *(const v8bf*)(LA + (mt * 16 + l15) * 64 + ksel * 32);
      v8bf ahi = *(const v8bf*)(LA + (mt * 16 + l15) * 64 + ksel * 32 + 16);
      v16bf a  = cat8(alo, ahi);
      v16bf b0 = *(const v16bf*)(LB + (ng * 32 + l15) * 64 + ksel * 32);
      v16bf b1 = *(const v16bf*)(LB + (ng * 32 + 16 + l15) * 64 + ksel * 32);
      acc0 = wmma_bf16(a, b0, acc0);
      acc1 = wmma_bf16(a, b1, acc1);

      __syncthreads();               // reads done before buffer is overwritten
    }
  }

  // ---- Epilogue: + bias + noise, softplus, leaky integrate, write h ----
  const int n0 = base_n + l15;
  const int n1 = base_n + 16 + l15;
  const float bi0 = bias[n0];
  const float bi1 = bias[n1];
#pragma unroll
  for (int v = 0; v < 8; ++v) {
    const int    m    = base_m + ksel * 8 + v;    // C/D layout: M = v + 8*(lane>>4)
    const size_t nrow = ((size_t)t * BATCH + m) * N_RNN;
    {
      float pre = acc0[v] + bi0 + noise[nrow + n0];
      float sp  = softplusf(pre);
      size_t hi = (size_t)m * N_RNN + n0;
      float hn  = 0.2f * sp + 0.8f * hf32[hi];
      hf32[hi] = hn;
      hcur[hi] = (__bf16)hn;
    }
    {
      float pre = acc1[v] + bi1 + noise[nrow + n1];
      float sp  = softplusf(pre);
      size_t hi = (size_t)m * N_RNN + n1;
      float hn  = 0.2f * sp + 0.8f * hf32[hi];
      hf32[hi] = hn;
      hcur[hi] = (__bf16)hn;
    }
  }
}

// ------------------------------- out kernel --------------------------------
// Grid: (BATCH/16, N_OUT_PAD/16) x 32 threads (one wave = one 16x16 tile).

__global__ __launch_bounds__(32, 1)
void RNN_out_kernel(const __bf16* __restrict__ hcur,   // (256, 1024)
                    const __bf16* __restrict__ wout,   // (48, 1024) padded
                    const float*  __restrict__ b_out,  // (33)
                    float*        __restrict__ out,    // (T, B, 33)
                    int t)
{
  const int lane = threadIdx.x & 31;
  const int l15  = lane & 15;
  const int ksel = lane >> 4;
  const int base_m = blockIdx.x * 16;
  const int base_n = blockIdx.y * 16;

  const __bf16* arow = hcur + (size_t)(base_m + l15) * N_RNN;
  const __bf16* brow = wout + (size_t)(base_n + l15) * N_RNN;
  v8f acc = {};
#pragma unroll 4
  for (int kk = 0; kk < N_RNN; kk += 32) {
    __builtin_prefetch(arow + kk + 256, 0, 3);   // global_prefetch_b8
    v8bf alo = *(const v8bf*)(arow + kk + ksel * 16);
    v8bf ahi = *(const v8bf*)(arow + kk + ksel * 16 + 8);
    v16bf a  = cat8(alo, ahi);
    v16bf b  = *(const v16bf*)(brow + kk + ksel * 16);
    acc = wmma_bf16(a, b, acc);
  }
  const int n = base_n + l15;
  if (n < N_OUT) {
    const float bo = b_out[n];
#pragma unroll
    for (int v = 0; v < 8; ++v) {
      const int m = base_m + ksel * 8 + v;
      out[((size_t)t * BATCH + m) * N_OUT + n] = acc[v] + bo;
    }
  }
}

// ------------------------------ host launcher ------------------------------

extern "C" void kernel_launch(void* const* d_in, const int* in_sizes, int n_in,
                              void* d_out, int out_size, void* d_ws, size_t ws_size,
                              hipStream_t stream) {
  (void)in_sizes; (void)n_in; (void)out_size; (void)ws_size;

  const float* x      = (const float*)d_in[0];
  const float* noise  = (const float*)d_in[1];
  const float* W_sens = (const float*)d_in[2];
  const float* b_sens = (const float*)d_in[3];
  const float* W_rule = (const float*)d_in[4];
  const float* W_rec  = (const float*)d_in[5];
  const float* b_rec  = (const float*)d_in[6];
  const float* W_out  = (const float*)d_in[7];
  const float* b_out  = (const float*)d_in[8];
  float* out = (float*)d_out;

  // Workspace carve-up (256B aligned), ~21 MB total.
  char* ws = (char*)d_ws;
  size_t off = 0;
  auto carve = [&](size_t bytes) {
    char* p = ws + off;
    off = (off + bytes + 255) & ~(size_t)255;
    return p;
  };
  __bf16* xbf  = (__bf16*)carve(sizeof(__bf16) * (size_t)T_STEPS * BATCH * K_IN);
  __bf16* wrec = (__bf16*)carve(sizeof(__bf16) * (size_t)N_RNN * N_RNN);
  __bf16* win  = (__bf16*)carve(sizeof(__bf16) * (size_t)N_RNN * K_IN);
  __bf16* wout = (__bf16*)carve(sizeof(__bf16) * (size_t)N_OUT_PAD * N_RNN);
  float*  bias = (float*)carve(sizeof(float) * N_RNN);
  float*  hf32 = (float*)carve(sizeof(float) * (size_t)BATCH * N_RNN);
  __bf16* hbf0 = (__bf16*)carve(sizeof(__bf16) * (size_t)BATCH * N_RNN);
  __bf16* hbf1 = (__bf16*)carve(sizeof(__bf16) * (size_t)BATCH * N_RNN);

  // ---- prep (re-run every call: deterministic, ws state not preserved) ----
  RNN_prep_x   <<<(T_STEPS * BATCH * K_IN) / 256, 256, 0, stream>>>(x, xbf);
  RNN_prep_wrec<<<(N_RNN * N_RNN) / 256,          256, 0, stream>>>(W_rec, wrec);
  RNN_prep_win <<<(N_RNN * K_IN) / 256,           256, 0, stream>>>(W_sens, W_rule, win);
  RNN_prep_wout<<<(N_OUT_PAD * N_RNN) / 256,      256, 0, stream>>>(W_out, wout);
  RNN_prep_misc<<<(BATCH * N_RNN) / 256,          256, 0, stream>>>(b_sens, b_rec, bias, hf32, hbf0);

  // ---- sequential scan: 256 steps, double-buffered bf16 hidden state ----
  const dim3 step_grid(N_RNN / 128, BATCH / 32);     // 8 x 8 workgroups
  const dim3 out_grid(BATCH / 16, N_OUT_PAD / 16);   // 16 x 3 waves
  for (int t = 0; t < T_STEPS; ++t) {
    const __bf16* hprev = (t & 1) ? hbf1 : hbf0;
    __bf16*       hcur  = (t & 1) ? hbf0 : hbf1;
    RNN_step_kernel<<<step_grid, 256, 0, stream>>>(xbf, wrec, win, bias, noise,
                                                   hf32, hprev, hcur, t);
    RNN_out_kernel<<<out_grid, 32, 0, stream>>>(hcur, wout, b_out, out, t);
  }
}
